// NeuralStackWithTools_2095944041027
// MI455X (gfx1250) — compile-verified, matrix operand order checked by reading, not run on hardware
//
#include <hip/hip_runtime.h>
#include <cstdint>
#include <cstddef>

// Problem sizes (fixed by setup_inputs)
#define B_    16
#define S_    2048
#define H_    1024
#define P_    2048   // 2*H
#define D_    16
#define T_    5
#define NROWS (B_ * S_)   // 32768

#define LN_EPS 1e-5f
#define EPS_   1e-8f

typedef __attribute__((ext_vector_type(16))) __bf16 bfx16;
typedef __attribute__((ext_vector_type(8)))  float  fx8;

union BfFrag { bfx16 v; uint4 q[2]; };

// Low 32 bits of a flat shared-space address are the LDS byte offset
// (the upper 32 bits only select the LDS aperture).
__device__ __forceinline__ unsigned lds_addr32(const void* p) {
  return (unsigned)(uintptr_t)p;
}

__device__ __forceinline__ void async_load_lds_b128(unsigned lds, const void* gaddr) {
  asm volatile("global_load_async_to_lds_b128 %0, %1, off"
               :: "v"(lds), "v"((unsigned long long)(uintptr_t)gaddr)
               : "memory");
}

// ---------------------------------------------------------------- utilities
__device__ __forceinline__ unsigned short f32_to_bf16_rne(float f) {
  unsigned int u = __float_as_uint(f);
  u += 0x7FFFu + ((u >> 16) & 1u);
  return (unsigned short)(u >> 16);
}

__global__ __launch_bounds__(256) void cvt_bf16_kernel(const float* __restrict__ in,
                                                       unsigned short* __restrict__ out,
                                                       size_t n) {
  size_t i = (size_t)blockIdx.x * 256 + threadIdx.x;
  if (i < n) out[i] = f32_to_bf16_rne(in[i]);
}

// ---------------------------------------------------------------- WMMA GEMM
// proj[m, n] = sum_k hidden_bf16[m, k] * W_bf16[n, k] + bias[n]
// M = 32768, N = 2048, K = 1024.
// Block = 8 waves (4 in M x 2 in N) -> 128x64 output tile per workgroup.
// 4-stage async global->LDS pipeline (issue depth 2, one barrier per K-step);
// fragments served to WMMA via ds_load_b128. K-loop fully unrolled so the
// accumulators stay as SSA chains (no cross-iteration v_mov copies).
#define GK_    32          // K elems per stage
#define LDA_   40          // padded LDS row stride in bf16 elems (80 B, 16B-aligned, conflict-free)
#define AR_    128
#define BR_    64
#define STG_E  (AR_ * LDA_ + BR_ * LDA_)   // 7680 shorts per stage (A then B)
#define STG_B  (STG_E * 2)                 // 15360 bytes per stage
#define NSTG_  4

__global__ __launch_bounds__(256) void gemm_wmma_kernel(
    const unsigned short* __restrict__ A,   // (M, K) bf16 bits
    const unsigned short* __restrict__ W,   // (N, K) bf16 bits
    const float* __restrict__ bias,         // (N)
    float* __restrict__ C)                  // (M, N) f32
{
  __shared__ __align__(16) unsigned short smem[NSTG_ * STG_E];   // 61440 B
  const int K = H_, N = P_;
  const int tid  = threadIdx.x;
  const int wave = tid >> 5;
  const int lane = tid & 31;
  const int l16  = lane & 15;
  const int kh   = lane >> 4;          // K-half-group of this lane

  const int wn = wave & 1, wm = wave >> 1;
  const int nBlk = blockIdx.x * 64;
  const int mBlk = blockIdx.y * 128;
  const int n0 = nBlk + wn * 32;
  const int m0 = mBlk + wm * 32;

  const unsigned ldsBase = lds_addr32(smem);

  fx8 c00 = {}, c01 = {}, c10 = {}, c11 = {};

  // Issue one stage of async global->LDS copies (3 b128 per thread).
  auto issue_stage = [&](int stage) {
    const unsigned sOff = (unsigned)(stage & (NSTG_ - 1)) * STG_B;
    const int k = stage * GK_;
#pragma unroll
    for (int j = 0; j < 2; ++j) {                      // A tile: 512 chunks of 16 B
      const int id = tid + j * 256;
      const int r = id >> 2, cch = id & 3;             // row, 16B-chunk in row
      async_load_lds_b128(ldsBase + sOff + (unsigned)(r * (LDA_ * 2) + cch * 16),
                          A + (size_t)(mBlk + r) * K + k + cch * 8);
    }
    {                                                  // B tile: 256 chunks of 16 B
      const int r = tid >> 2, cch = tid & 3;
      async_load_lds_b128(ldsBase + sOff + (unsigned)(AR_ * LDA_ * 2 + r * (LDA_ * 2) + cch * 16),
                          W + (size_t)(nBlk + r) * K + k + cch * 8);
    }
  };

  const int NIT = K / GK_;                             // 32
  issue_stage(0);
  issue_stage(1);

#pragma unroll
  for (int it = 0; it < NIT; ++it) {
    if (it + 2 < NIT) {
      issue_stage(it + 2);
      asm volatile("s_wait_asynccnt 0x6" ::: "memory");  // stage `it` landed
    } else if (it + 1 < NIT) {
      asm volatile("s_wait_asynccnt 0x3" ::: "memory");
    } else {
      asm volatile("s_wait_asynccnt 0x0" ::: "memory");
    }
    __syncthreads();   // all waves' stage-`it` data visible; buffer being
                       // overwritten (it+2 mod 4) was last read at it-2.

    const unsigned short* sA = smem + (it & (NSTG_ - 1)) * STG_E;
    const unsigned short* sB = sA + AR_ * LDA_;

    // Fragment reads from LDS (ds_load_b128), per ISA bf16 A/B layouts.
    const unsigned short* a0p = sA + (wm * 32 +      l16) * LDA_ + kh * 8;
    const unsigned short* a1p = sA + (wm * 32 + 16 + l16) * LDA_ + kh * 8;
    const unsigned short* b0p = sB + (wn * 32 +      l16) * LDA_ + kh * 8;
    const unsigned short* b1p = sB + (wn * 32 + 16 + l16) * LDA_ + kh * 8;

    BfFrag a0, a1, b0, b1;
    a0.q[0] = *(const uint4*)(a0p);  a0.q[1] = *(const uint4*)(a0p + 16);
    a1.q[0] = *(const uint4*)(a1p);  a1.q[1] = *(const uint4*)(a1p + 16);
    b0.q[0] = *(const uint4*)(b0p);  b0.q[1] = *(const uint4*)(b0p + 16);
    b1.q[0] = *(const uint4*)(b1p);  b1.q[1] = *(const uint4*)(b1p + 16);

    c00 = __builtin_amdgcn_wmma_f32_16x16x32_bf16(false, a0.v, false, b0.v, (short)0, c00, false, false);
    c01 = __builtin_amdgcn_wmma_f32_16x16x32_bf16(false, a0.v, false, b1.v, (short)0, c01, false, false);
    c10 = __builtin_amdgcn_wmma_f32_16x16x32_bf16(false, a1.v, false, b0.v, (short)0, c10, false, false);
    c11 = __builtin_amdgcn_wmma_f32_16x16x32_bf16(false, a1.v, false, b1.v, (short)0, c11, false, false);
  }

  // D layout: lane L -> col N=L%16, VGPR r -> row M = r + (L/16)*8
  const float bias0 = bias[n0 +      l16];
  const float bias1 = bias[n0 + 16 + l16];
#pragma unroll
  for (int r = 0; r < 8; ++r) {
    const int mr0 = m0 +      kh * 8 + r;
    const int mr1 = m0 + 16 + kh * 8 + r;
    C[(size_t)mr0 * N + n0 +      l16] = c00[r] + bias0;
    C[(size_t)mr0 * N + n0 + 16 + l16] = c01[r] + bias1;
    C[(size_t)mr1 * N + n0 +      l16] = c10[r] + bias0;
    C[(size_t)mr1 * N + n0 + 16 + l16] = c11[r] + bias1;
  }
}

// ---------------------------------------------------------------- logits + softmax
__global__ __launch_bounds__(256) void probs_kernel(
    const float* __restrict__ hidden, const float* __restrict__ proj,
    const float* __restrict__ W_op, const float* __restrict__ b_op,
    const float* __restrict__ W_tool, const float* __restrict__ b_tool,
    float* __restrict__ op_probs, float* __restrict__ tool_w_out)
{
  __shared__ float red[256 * 8];
  const size_t row = blockIdx.x;
  const int t = threadIdx.x;
  const float* hrow = hidden + row * H_;
  const float* arow = proj   + row * (size_t)P_;     // h_all = proj[:, 0:H]

  float acc[8] = {0,0,0,0,0,0,0,0};
  for (int i = t; i < H_; i += 256) {
    const float hv = hrow[i];
    const float ha = arow[i];
    acc[0] += hv * W_op[0 * H_ + i];
    acc[1] += hv * W_op[1 * H_ + i];
    acc[2] += hv * W_op[2 * H_ + i];
    acc[3] += ha * W_tool[0 * H_ + i];
    acc[4] += ha * W_tool[1 * H_ + i];
    acc[5] += ha * W_tool[2 * H_ + i];
    acc[6] += ha * W_tool[3 * H_ + i];
    acc[7] += ha * W_tool[4 * H_ + i];
  }
#pragma unroll
  for (int j = 0; j < 8; ++j) red[t * 8 + j] = acc[j];
  __syncthreads();
  for (int off = 128; off > 0; off >>= 1) {
    if (t < off) {
#pragma unroll
      for (int j = 0; j < 8; ++j) red[t * 8 + j] += red[(t + off) * 8 + j];
    }
    __syncthreads();
  }
  if (t == 0) {
    float l3[3], l5[5];
    float mx = -1e30f;
    for (int j = 0; j < 3; ++j) { l3[j] = red[j] + b_op[j]; mx = fmaxf(mx, l3[j]); }
    float sm = 0.f;
    for (int j = 0; j < 3; ++j) { l3[j] = expf(l3[j] - mx); sm += l3[j]; }
    for (int j = 0; j < 3; ++j) op_probs[row * 3 + j] = l3[j] / sm;
    mx = -1e30f;
    for (int j = 0; j < 5; ++j) { l5[j] = red[3 + j] + b_tool[j]; mx = fmaxf(mx, l5[j]); }
    sm = 0.f;
    for (int j = 0; j < 5; ++j) { l5[j] = expf(l5[j] - mx); sm += l5[j]; }
    for (int j = 0; j < 5; ++j) tool_w_out[row * T_ + j] = l5[j] / sm;
  }
}

// ---------------------------------------------------------------- pointer trajectory scan
__global__ __launch_bounds__(32) void pointer_scan_kernel(
    const float* __restrict__ op_probs,   // (B*S, 3)
    float* __restrict__ p_seq,            // (B*S, 16)
    float* __restrict__ w_seq,            // (B*S, 16)
    float* __restrict__ out_pointer_f)    // (B, 16)
{
  const int b = blockIdx.x;
  const int lane = threadIdx.x;
  const int d = lane & 15;                // lanes 16..31 mirror lanes 0..15
  float p = (d == 0) ? 1.0f : 0.0f;

  for (int s = 0; s < S_; ++s) {
    const size_t row = (size_t)b * S_ + s;
    const float push = op_probs[row * 3 + 0];
    const float pop  = op_probs[row * 3 + 1];
    const float nop  = op_probs[row * 3 + 2];
    const float p_push = __shfl(p, (lane + 15) & 15, 32);   // p[(d-1)%16]
    const float p_pop  = __shfl(p, (lane + 1)  & 15, 32);   // p[(d+1)%16]
    const float w  = push * p_push;
    float np = w + pop * p_pop + nop * p;
    float q  = np * np;                                     // np^5
    np = q * q * np;
    float sm = np;
    sm += __shfl_xor(sm, 1, 32);
    sm += __shfl_xor(sm, 2, 32);
    sm += __shfl_xor(sm, 4, 32);
    sm += __shfl_xor(sm, 8, 32);
    const float inv = 1.0f / (sm + EPS_);
    p = np * inv;
    if (lane < 16) {
      p_seq[row * D_ + d] = p;
      w_seq[row * D_ + d] = w;
    }
  }
  if (lane < 16) out_pointer_f[b * D_ + d] = p;
}

// ---------------------------------------------------------------- parallel stack scan
// Each thread owns one (b, h): 16 depth cells in registers. Sequential over S
// with w/p trajectories async-staged into LDS tiles. Fuses tool mix + residual
// and overwrites the consumed v_all slot of proj with `combined`.
#define TS_ 64
__global__ __launch_bounds__(256) void stack_scan_kernel(
    float* proj,                            // (B*S, 2048): [:,0:H]=h_all, [:,H:2H]=v_all -> combined
    const float* __restrict__ hidden,       // (B*S, H)
    const float* __restrict__ p_seq,        // (B*S, 16)
    const float* __restrict__ w_seq,        // (B*S, 16)
    const float* __restrict__ tool_w,       // (B*S, 5)
    float* __restrict__ out_stack_f)        // (B, D, H)
{
  __shared__ __align__(16) float sp[TS_ * D_];
  __shared__ __align__(16) float sw[TS_ * D_];
  const int b = blockIdx.y;
  const int h = blockIdx.x * 256 + threadIdx.x;

  float x[D_];
#pragma unroll
  for (int d = 0; d < D_; ++d) x[d] = 0.0f;

  for (int s0 = 0; s0 < S_; s0 += TS_) {
    __syncthreads();                 // previous tile fully consumed
    {
      const size_t base = ((size_t)b * S_ + s0) * D_ + (size_t)threadIdx.x * 4;
      async_load_lds_b128(lds_addr32(sp) + threadIdx.x * 16, p_seq + base);
      async_load_lds_b128(lds_addr32(sw) + threadIdx.x * 16, w_seq + base);
      asm volatile("s_wait_asynccnt 0x0" ::: "memory");
    }
    __syncthreads();

    for (int si = 0; si < TS_; ++si) {
      const size_t row = (size_t)b * S_ + (s0 + si);
      const float v = proj[row * P_ + H_ + h];
      float top = 0.0f;
#pragma unroll
      for (int d = 0; d < D_; ++d) {
        const float w = sw[si * D_ + d];
        x[d] += w * (v - x[d]);                 // new_stack
        top  += x[d] * sp[si * D_ + d];         // read-out with normalized new_p
      }
      const float* tw = tool_w + row * T_;
      const float t0 = tw[0], t1 = tw[1], t2 = tw[2], t3 = tw[3], t4 = tw[4];
      const float rl = fmaxf(top, 0.0f);
      const float ge = 0.5f * top * (1.0f + erff(top * 0.70710678118654752f));
      const float th = tanhf(top);
      const float sg = 1.0f / (1.0f + expf(-top));
      const float processed = rl * t0 + ge * t1 + th * t2 + sg * t3 + top * t4;
      const float comb = proj[row * P_ + h] + processed + hidden[row * H_ + h];
      proj[row * P_ + H_ + h] = comb;           // overwrite consumed v_all slot
    }
  }
#pragma unroll
  for (int d = 0; d < D_; ++d)
    out_stack_f[(size_t)b * D_ * H_ + (size_t)d * H_ + h] = x[d];
}

// ---------------------------------------------------------------- LayerNorm
__global__ __launch_bounds__(256) void ln_kernel(
    const float* __restrict__ proj,        // combined lives at [:, H:2H]
    const float* __restrict__ gamma, const float* __restrict__ beta,
    float* __restrict__ out)               // (B*S, H)
{
  __shared__ float rsum[256];
  __shared__ float rsq[256];
  const size_t row = blockIdx.x;
  const int t = threadIdx.x;
  const float* c = proj + row * (size_t)P_ + H_;

  float s = 0.f, sq = 0.f;
  for (int i = t; i < H_; i += 256) {
    const float v = c[i];
    s += v; sq += v * v;
  }
  rsum[t] = s; rsq[t] = sq;
  __syncthreads();
  for (int off = 128; off > 0; off >>= 1) {
    if (t < off) { rsum[t] += rsum[t + off]; rsq[t] += rsq[t + off]; }
    __syncthreads();
  }
  const float mu  = rsum[0] * (1.0f / H_);
  const float var = rsq[0] * (1.0f / H_) - mu * mu;
  const float inv = rsqrtf(var + LN_EPS);
  for (int i = t; i < H_; i += 256)
    out[row * H_ + i] = (c[i] - mu) * inv * gamma[i] + beta[i];
}

// ---------------------------------------------------------------- launch
extern "C" void kernel_launch(void* const* d_in, const int* in_sizes, int n_in,
                              void* d_out, int out_size, void* d_ws, size_t ws_size,
                              hipStream_t stream) {
  const float* hidden = (const float*)d_in[0];
  const float* W_proj = (const float*)d_in[1];
  const float* b_proj = (const float*)d_in[2];
  const float* W_op   = (const float*)d_in[3];
  const float* b_op   = (const float*)d_in[4];
  const float* W_tool = (const float*)d_in[5];
  const float* b_tool = (const float*)d_in[6];
  const float* ln_g   = (const float*)d_in[7];
  const float* ln_b   = (const float*)d_in[8];

  float* out_final   = (float*)d_out;                                  // (B,S,H)
  float* out_stack   = out_final   + (size_t)NROWS * H_;               // (B,D,H)
  float* out_pointer = out_stack   + (size_t)B_ * D_ * H_;             // (B,D)
  float* out_toolw   = out_pointer + (size_t)B_ * D_;                  // (B,S,T)

  char* ws = (char*)d_ws;
  unsigned short* hbf = (unsigned short*)(ws);                         //  64 MB
  unsigned short* wbf = (unsigned short*)(ws + 67108864ull);           //   4 MB
  float* proj         = (float*)(ws + 71303168ull);                    // 256 MB
  float* op_probs     = (float*)(ws + 339738624ull);                   // 384 KB
  float* p_seq        = (float*)(ws + 340131840ull);                   //   2 MB
  float* w_seq        = (float*)(ws + 342228992ull);                   //   2 MB

  {
    const size_t n = (size_t)NROWS * H_;
    cvt_bf16_kernel<<<(unsigned)((n + 255) / 256), 256, 0, stream>>>(hidden, hbf, n);
  }
  {
    const size_t n = (size_t)P_ * H_;
    cvt_bf16_kernel<<<(unsigned)((n + 255) / 256), 256, 0, stream>>>(W_proj, wbf, n);
  }

  gemm_wmma_kernel<<<dim3(P_ / 64, NROWS / 128), 256, 0, stream>>>(hbf, wbf, b_proj, proj);

  probs_kernel<<<NROWS, 256, 0, stream>>>(hidden, proj, W_op, b_op, W_tool, b_tool,
                                          op_probs, out_toolw);

  pointer_scan_kernel<<<B_, 32, 0, stream>>>(op_probs, p_seq, w_seq, out_pointer);

  stack_scan_kernel<<<dim3(H_ / 256, B_), 256, 0, stream>>>(proj, hidden, p_seq, w_seq,
                                                            out_toolw, out_stack);

  ln_kernel<<<NROWS, 256, 0, stream>>>(proj, ln_g, ln_b, out_final);
}
